// OFTLinearLayer_38354057954038
// MI455X (gfx1250) — matrix operand order repaired and checked by reading, not compiled, not run on hardware
//
#include <hip/hip_runtime.h>
#include <hip/hip_bf16.h>

// ---------------------------------------------------------------------------
// OFT linear layer, MI455X (gfx1250, wave32, WMMA + async LDS staging).
//   x: [32,512,1024] f32   Q: [4,256,256] f32
//   W: [4096,1024]   f32   b: [4096] f32
//   out: [16384,4096] f32
// Pipeline: bf16-convert x -> build [I-S | I+S] -> Gauss-Jordan (rot^T) ->
//           filter GEMM (bf16 [4096][1024]) -> WMMA main GEMM (f32 acc),
//           staged via global_load_async_to_lds_b128 (ASYNCcnt) + double buffer.
// ---------------------------------------------------------------------------

typedef __attribute__((ext_vector_type(16))) __bf16    v16bf;
typedef __attribute__((ext_vector_type(8)))  float     v8f;
typedef __attribute__((ext_vector_type(4)))  float     v4f;
typedef __attribute__((ext_vector_type(4)))  unsigned  v4u;

static __device__ __forceinline__ unsigned short f2bf(float f) {
  unsigned u = __builtin_bit_cast(unsigned, f);
  unsigned r = u + 0x7FFFu + ((u >> 16) & 1u);   // round-to-nearest-even
  return (unsigned short)(r >> 16);
}

// ---------------------------------------------------------------- kernel 1 --
__global__ __launch_bounds__(256)
void x_to_bf16(const float* __restrict__ x, unsigned short* __restrict__ xb, int n4) {
  int i = blockIdx.x * 256 + threadIdx.x;
  if (i < n4) {
    v4f v = __builtin_nontemporal_load(&((const v4f*)x)[i]);
    uint2 o;
    o.x = (unsigned)f2bf(v.x) | ((unsigned)f2bf(v.y) << 16);
    o.y = (unsigned)f2bf(v.z) | ((unsigned)f2bf(v.w) << 16);
    ((uint2*)xb)[i] = o;
  }
}

// ---------------------------------------------------------------- kernel 2 --
// aug[r] = [ (I - S) | (I + S) ]  (256 x 512), S = 0.5*(Q - Q^T).
// Since (I+S)^T = I-S, GJ on [B|B^T] yields [I | rot^T].
__global__ __launch_bounds__(256)
void build_aug(const float* __restrict__ Q, float* __restrict__ aug) {
  int bid = blockIdx.x;
  int r = bid >> 8, i = bid & 255;
  int j = threadIdx.x;
  const float* Qr = Q + (long)r * 256 * 256;
  float s    = 0.5f * (Qr[i * 256 + j] - Qr[j * 256 + i]);
  float diag = (i == j) ? 1.0f : 0.0f;
  float* row = aug + ((long)r * 256 + i) * 512;
  row[j]       = diag - s;   // B   = I - S
  row[256 + j] = diag + s;   // B^T = I + S
}

// ---------------------------------------------------------------- kernel 3 --
// In-place Gauss-Jordan, one workgroup per block matrix. I-S has unit diagonal
// and is well-conditioned (S skew), so no pivoting. Pivot row cached in LDS.
__global__ __launch_bounds__(256)
void gauss_jordan(float* __restrict__ aug) {
  float* Mt = aug + (long)blockIdx.x * 256 * 512;
  __shared__ float prow[512];
  int tid = threadIdx.x;
  for (int p = 0; p < 256; ++p) {
    float inv = 1.0f / Mt[p * 512 + p];
    __syncthreads();
    for (int j = tid; j < 512; j += 256) {
      float v = Mt[p * 512 + j] * inv;
      prow[j] = v;
      Mt[p * 512 + j] = v;
    }
    __syncthreads();
    if (tid != p) {
      float f = Mt[tid * 512 + p];
      #pragma unroll 8
      for (int j = 0; j < 512; ++j)
        Mt[tid * 512 + j] -= f * prow[j];
    }
    __syncthreads();
  }
}

// ---------------------------------------------------------------- kernel 4 --
// filtT[o][i] = sum_j W[o][r*256+j] * rotT[r][j][ii]   (i = r*256+ii)
// rotT lives in aug[r][j][256+ii]. Output bf16 [4096][1024] (K-contiguous for
// the main GEMM's B fragments). fp32 LDS-tiled 32x32, 2x2 per thread.
__global__ __launch_bounds__(256)
void filt_gemm(const float* __restrict__ aug, const float* __restrict__ W,
               unsigned short* __restrict__ filtT) {
  int it = blockIdx.x;            // 0..31  (i tiles)
  int ot = blockIdx.y;            // 0..127 (o tiles)
  int r   = it >> 3;
  int ii0 = (it & 7) * 32;
  int o0  = ot * 32;
  __shared__ __align__(16) float Ws[32 * 36];
  __shared__ __align__(16) float Rs[32 * 36];
  int tid  = threadIdx.x;
  int row  = tid >> 3;
  int col4 = (tid & 7) * 4;
  int tx   = tid & 15, ty = tid >> 4;
  float acc[2][2] = {{0.f, 0.f}, {0.f, 0.f}};
  const float* augr = aug + (long)r * 256 * 512;
  for (int j0 = 0; j0 < 256; j0 += 32) {
    float4 wv = *(const float4*)&W[(long)(o0 + row) * 1024 + r * 256 + j0 + col4];
    float4 rv = *(const float4*)&augr[(long)(j0 + row) * 512 + 256 + ii0 + col4];
    *(float4*)&Ws[row * 36 + col4] = wv;
    *(float4*)&Rs[row * 36 + col4] = rv;
    __syncthreads();
    #pragma unroll
    for (int j = 0; j < 32; ++j) {
      float a0 = Ws[(ty * 2 + 0) * 36 + j];
      float a1 = Ws[(ty * 2 + 1) * 36 + j];
      float b0 = Rs[j * 36 + tx * 2 + 0];
      float b1 = Rs[j * 36 + tx * 2 + 1];
      acc[0][0] += a0 * b0; acc[0][1] += a0 * b1;
      acc[1][0] += a1 * b0; acc[1][1] += a1 * b1;
    }
    __syncthreads();
  }
  #pragma unroll
  for (int d = 0; d < 2; ++d)
    #pragma unroll
    for (int e = 0; e < 2; ++e)
      filtT[(long)(o0 + ty * 2 + d) * 1024 + r * 256 + ii0 + tx * 2 + e] =
          f2bf(acc[d][e]);
}

// ---------------------------------------------------------------- kernel 5 --
// out[m][n] = sum_k xb[m][k]*filtT[n][k] + bias[n], via v_wmma_f32_16x16x32_bf16.
// Block tile 128(M) x 256(N), BK=32, 8 waves in 4(M)x2(N) grid, 2x8 WMMA
// tiles/wave. Tiles staged with global_load_async_to_lds_b128 (double buffer,
// ASYNCcnt-synchronized). NT stores in the epilogue keep L2 for A/B reuse.
#define BM 128
#define BN 256
#define BK 32
#define LDH 40   // LDS row stride in halfs (80 B: b128-aligned, spreads banks)
#define KITERS 32  // 1024 / BK

union FragBF { v16bf v; v4u q[2]; };

__global__ __launch_bounds__(256)
void oft_gemm(const unsigned short* __restrict__ A,   // xb   [16384][1024]
              const unsigned short* __restrict__ B,   // filtT[4096][1024]
              const float* __restrict__ bias,
              float* __restrict__ C) {
  const int K = 1024, N = 4096;
  __shared__ __align__(16) unsigned short As[2][BM * LDH];   // 2 x 10 KB
  __shared__ __align__(16) unsigned short Bs[2][BN * LDH];   // 2 x 20 KB

  int tid  = threadIdx.x;
  int bx   = blockIdx.x;          // N tiles (16)
  int by   = blockIdx.y;          // M tiles (128)
  int lane = tid & 31, wave = tid >> 5;
  int wm = wave >> 1;             // 0..3 -> wave M offset wm*32
  int wn = wave & 1;              // 0..1 -> wave N offset wn*128
  int lhalf = lane >> 4;          // K-half selector per ISA bf16 layouts
  int lrow  = lane & 15;

  // staging map: A -> thread covers 32 contiguous B of row lr; B -> 64 B of row tid
  int lr = tid >> 1;
  int lc = (tid & 1) * 16;        // halfs
  unsigned long long gA = (unsigned long long)(A + (long)(by * BM + lr) * K + lc);
  unsigned long long gB = (unsigned long long)(B + (long)(bx * BN + tid) * K);
  unsigned ldsA[2], ldsB[2];
  ldsA[0] = (unsigned)(unsigned long long)&As[0][lr * LDH + lc];
  ldsA[1] = (unsigned)(unsigned long long)&As[1][lr * LDH + lc];
  ldsB[0] = (unsigned)(unsigned long long)&Bs[0][tid * LDH];
  ldsB[1] = (unsigned)(unsigned long long)&Bs[1][tid * LDH];

  v8f acc[2][8] = {};

  auto prefetch = [&](int k0, int buf) {
    unsigned long long ga = gA + (unsigned long long)k0 * 2;
    unsigned long long gb = gB + (unsigned long long)k0 * 2;
    // INST_OFFSET is added to BOTH the LDS dest and global src (ISA 10.7),
    // so one (lds,global) register pair serves all chunks of this thread.
    asm volatile(
        "global_load_async_to_lds_b128 %0, %2, off\n\t"
        "global_load_async_to_lds_b128 %0, %2, off offset:16\n\t"
        "global_load_async_to_lds_b128 %1, %3, off\n\t"
        "global_load_async_to_lds_b128 %1, %3, off offset:16\n\t"
        "global_load_async_to_lds_b128 %1, %3, off offset:32\n\t"
        "global_load_async_to_lds_b128 %1, %3, off offset:48"
        :: "v"(ldsA[buf]), "v"(ldsB[buf]), "v"(ga), "v"(gb)
        : "memory");
  };

  prefetch(0, 0);

  #pragma unroll 2
  for (int it = 0; it < KITERS; ++it) {
    int buf = it & 1;
    if (it + 1 < KITERS) {
      prefetch((it + 1) * BK, buf ^ 1);
      // 6 ops of stage it+1 outstanding; async loads complete in order, so
      // ASYNCcnt <= 6 guarantees stage it has fully landed in LDS.
      asm volatile("s_wait_asynccnt 0x6" ::: "memory");
    } else {
      asm volatile("s_wait_asynccnt 0x0" ::: "memory");
    }
    __syncthreads();

    const unsigned short* as = &As[buf][0];
    const unsigned short* bs = &Bs[buf][0];

    // A fragment (16x32 bf16): lanes<16 hold K{0..7,16..23}, lanes>=16 K{8..15,24..31}
    FragBF af[2];
    #pragma unroll
    for (int tm = 0; tm < 2; ++tm) {
      int m = wm * 32 + tm * 16 + lrow;
      af[tm].q[0] = *(const v4u*)&as[m * LDH + lhalf * 8];
      af[tm].q[1] = *(const v4u*)&as[m * LDH + 16 + lhalf * 8];
    }
    // B fragment (32x16 bf16): column n=lrow, same K order as A:
    // lanes<16 hold K{0..7,16..23}, lanes>=16 K{8..15,24..31}
    FragBF bfr[8];
    #pragma unroll
    for (int tn = 0; tn < 8; ++tn) {
      int n = wn * 128 + tn * 16 + lrow;
      bfr[tn].q[0] = *(const v4u*)&bs[n * LDH + lhalf * 8];
      bfr[tn].q[1] = *(const v4u*)&bs[n * LDH + 16 + lhalf * 8];
    }

    #pragma unroll
    for (int tm = 0; tm < 2; ++tm)
      #pragma unroll
      for (int tn = 0; tn < 8; ++tn)
        acc[tm][tn] = __builtin_amdgcn_wmma_f32_16x16x32_bf16(
            false, af[tm].v, false, bfr[tn].v, (short)0, acc[tm][tn],
            false, false);

    __syncthreads();   // protect buf from stage it+2's async overwrite
  }

  // epilogue: C/D layout -> lane L, VGPR v maps to (m = base+(L>>4)*8+v, n = base+(L&15)).
  // Output is written once and never re-read: NT stores preserve L2 for A/B tiles.
  #pragma unroll
  for (int tm = 0; tm < 2; ++tm) {
    #pragma unroll
    for (int tn = 0; tn < 8; ++tn) {
      int n = bx * BN + wn * 128 + tn * 16 + lrow;
      float bv = bias[n];
      int mbase = by * BM + wm * 32 + tm * 16 + lhalf * 8;
      #pragma unroll
      for (int v = 0; v < 8; ++v)
        __builtin_nontemporal_store(acc[tm][tn][v] + bv,
                                    &C[(long)(mbase + v) * N + n]);
    }
  }
}

// ---------------------------------------------------------------------------
extern "C" void kernel_launch(void* const* d_in, const int* in_sizes, int n_in,
                              void* d_out, int out_size, void* d_ws, size_t ws_size,
                              hipStream_t stream) {
  const float* x    = (const float*)d_in[0];   // 32*512*1024
  const float* Q    = (const float*)d_in[1];   // 4*256*256
  const float* W    = (const float*)d_in[2];   // 4096*1024
  const float* bias = (const float*)d_in[3];   // 4096
  float* out = (float*)d_out;                  // 16384*4096

  char* ws = (char*)d_ws;
  float*          aug   = (float*)ws;                                   //  2 MB
  unsigned short* xb    = (unsigned short*)(ws + 2097152);              // 32 MB
  unsigned short* filtT = (unsigned short*)(ws + 2097152 + 33554432);   //  8 MB

  x_to_bf16   <<<16384,         256, 0, stream>>>(x, xb, 4194304);
  build_aug   <<<1024,          256, 0, stream>>>(Q, aug);
  gauss_jordan<<<4,             256, 0, stream>>>(aug);
  filt_gemm   <<<dim3(32, 128), 256, 0, stream>>>(aug, W, filtT);
  oft_gemm    <<<dim3(16, 128), 256, 0, stream>>>(xb, filtT, bias, out);
}
